// LinearAttention_33217277067419
// MI455X (gfx1250) — compile-verified, matrix operand order checked by reading, not run on hardware
//
#include <hip/hip_runtime.h>

typedef __attribute__((ext_vector_type(16))) _Float16 v16h;
typedef __attribute__((ext_vector_type(4)))  _Float16 v4h;
typedef __attribute__((ext_vector_type(8)))  float    v8f;
typedef __attribute__((ext_vector_type(4)))  float    v4f;

constexpr int BATCH = 16;
constexpr int CDIM  = 256;
constexpr int NHEAD = 4;
constexpr int DHEAD = 32;
constexpr int HID   = 128;   // NHEAD * DHEAD
constexpr int NQKV  = 384;   // 3 * HID
constexpr int NPIX  = 4096;  // 64 * 64
constexpr float QSCALE = 0.17677669529663687f; // 32^-0.5
constexpr float INV_N  = 1.0f / 4096.0f;
constexpr float LN_EPS = 1e-5f;

__device__ __forceinline__ v8f vzero8() { v8f z = {}; return z; }

__device__ __forceinline__ v8f wmma_f16(v16h a, v16h b, v8f c) {
  return __builtin_amdgcn_wmma_f32_16x16x32_f16(false, a, false, b, (short)0, c,
                                                false, false);
}

// Fragment from row-major f32 memory. Per ISA 7.12.2: lane holds
// K = {kh*8+j, 16+kh*8+j}, kh = lane>>4.
__device__ __forceinline__ v16h frag_g_row(const float* __restrict__ p, int kh) {
  v16h f;
  const float* p0 = p + kh * 8;
  const float* p1 = p + 16 + kh * 8;
#pragma unroll
  for (int j = 0; j < 8; ++j) f[j] = (_Float16)p0[j];
#pragma unroll
  for (int j = 0; j < 8; ++j) f[8 + j] = (_Float16)p1[j];
  return f;
}

// Same, but applies softmax normalization exp(x - mx) * rcp on the fly.
__device__ __forceinline__ v16h frag_g_row_sm(const float* __restrict__ p, int kh,
                                              float mx, float rcp) {
  v16h f;
  const float* p0 = p + kh * 8;
  const float* p1 = p + 16 + kh * 8;
#pragma unroll
  for (int j = 0; j < 8; ++j) f[j] = (_Float16)(__expf(p0[j] - mx) * rcp);
#pragma unroll
  for (int j = 0; j < 8; ++j) f[8 + j] = (_Float16)(__expf(p1[j] - mx) * rcp);
  return f;
}

// Fragment from LDS halves, K-contiguous -> 2x ds_load_b128.
__device__ __forceinline__ v16h frag_lds_row(const _Float16* row, int kh) {
  v16h f;
  const _Float16* p0 = row + kh * 8;
  const _Float16* p1 = row + 16 + kh * 8;
#pragma unroll
  for (int j = 0; j < 8; ++j) f[j] = p0[j];
#pragma unroll
  for (int j = 0; j < 8; ++j) f[8 + j] = p1[j];
  return f;
}

// ---------------------------------------------------------------------------
// K1: qkv[b, 0:384, n] = w_qkv[384,256] @ x[b, 256, n];  v rows scaled by 1/n
// block = 256 threads (8 waves), tile = 128(M) x 128(N), K chunk = 32,
// double-buffered LDS (one barrier/chunk; loads overlap WMMA).
// ---------------------------------------------------------------------------
__global__ __launch_bounds__(256) void k_qkv_gemm(const float* __restrict__ x,
                                                  const float* __restrict__ w_qkv,
                                                  float* __restrict__ qkv) {
  __shared__ _Float16 Asl[2][128][40];  // [buf][m][k]
  __shared__ _Float16 Bt[2][128][40];   // [buf][n][k] (transposed x tile)
  const int n0 = blockIdx.x * 128;
  const int m0 = blockIdx.y * 128;
  const int b  = blockIdx.z;
  const int t = threadIdx.x;
  const int wave = t >> 5, lane = t & 31;
  const int m = lane & 15, kh = lane >> 4;
  const int wm = wave >> 2;  // 0..1: M offset wm*64
  const int wn = wave & 3;   // 0..3: N offset wn*32
  const float* xb = x + (size_t)b * CDIM * NPIX;
  float* qb = qkv + (size_t)b * NQKV * NPIX;
  // m0 in {0,128,256}; v-boundary (256) aligned with tile -> block-uniform
  const float vscale = (m0 >= 2 * HID) ? INV_N : 1.0f;

  v8f cc[4][2];
#pragma unroll
  for (int i = 0; i < 4; ++i)
#pragma unroll
    for (int j = 0; j < 2; ++j) cc[i][j] = vzero8();

  auto stage = [&](int buf, int kk) {
    // A: 128 rows x 32 k (w_qkv row-major); float4 loads, packed v4h stores
    {
      const int r0 = t >> 3;        // 0..31
      const int c4 = (t & 7) * 4;   // k offset
#pragma unroll
      for (int i = 0; i < 4; ++i) {
        const int r = r0 + i * 32;
        v4f d = *(const v4f*)&w_qkv[(m0 + r) * CDIM + kk + c4];
        v4h hh;
#pragma unroll
        for (int j = 0; j < 4; ++j) hh[j] = (_Float16)d[j];
        *(v4h*)&Asl[buf][r][c4] = hh;
      }
    }
    // B: 32 k x 128 n from x (coalesced float4), transpose into Bt[n][k]
    {
      const int r0 = t >> 5;         // 0..7 (k)
      const int c4 = (t & 31) * 4;   // n offset
#pragma unroll
      for (int i = 0; i < 4; ++i) {
        const int r = r0 + i * 8;    // k 0..31
        v4f d = *(const v4f*)&xb[(size_t)(kk + r) * NPIX + n0 + c4];
#pragma unroll
        for (int j = 0; j < 4; ++j) Bt[buf][c4 + j][r] = (_Float16)d[j];
      }
    }
  };

  stage(0, 0);
  __syncthreads();
  for (int kk = 0; kk < CDIM; kk += 32) {
    const int cur = (kk >> 5) & 1;
    if (kk + 32 < CDIM) stage(cur ^ 1, kk + 32);  // overlaps with WMMAs below
    v16h bf[2];
#pragma unroll
    for (int nt = 0; nt < 2; ++nt)
      bf[nt] = frag_lds_row(&Bt[cur][wn * 32 + nt * 16 + m][0], kh);
#pragma unroll
    for (int mt = 0; mt < 4; ++mt) {
      v16h a = frag_lds_row(&Asl[cur][wm * 64 + mt * 16 + m][0], kh);
#pragma unroll
      for (int nt = 0; nt < 2; ++nt) cc[mt][nt] = wmma_f16(a, bf[nt], cc[mt][nt]);
    }
    __syncthreads();
  }

  const int mo = kh * 8;
#pragma unroll
  for (int mt = 0; mt < 4; ++mt)
#pragma unroll
    for (int nt = 0; nt < 2; ++nt)
#pragma unroll
      for (int g = 0; g < 8; ++g) {
        const int mg = m0 + wm * 64 + mt * 16 + g + mo;
        const int ng = n0 + wn * 32 + nt * 16 + m;
        qb[(size_t)mg * NPIX + ng] = cc[mt][nt][g] * vscale;
      }
}

// ---------------------------------------------------------------------------
// K2: softmax stats (row max, 1/sumexp) over n=4096 per k-row. No rewrite;
// normalization is folded into K3's fragment conversion.
// ---------------------------------------------------------------------------
__global__ __launch_bounds__(256) void k_softmax_stats(const float* __restrict__ qkv,
                                                       float* __restrict__ kstats) {
  const int row = blockIdx.x;              // 0..2047  (= b*128 + h*32 + d)
  const int b = row >> 7, r = row & 127;
  const float* p = qkv + (size_t)b * NQKV * NPIX + (size_t)(HID + r) * NPIX;
  __shared__ float redmax[8];
  __shared__ float redsum[8];
  const int t = threadIdx.x, lane = t & 31, w = t >> 5;

  float mx = -3.4e38f;
  for (int i = t; i < NPIX; i += 256) mx = fmaxf(mx, p[i]);
#pragma unroll
  for (int o = 16; o > 0; o >>= 1) mx = fmaxf(mx, __shfl_xor(mx, o, 32));
  if (lane == 0) redmax[w] = mx;
  __syncthreads();
  if (t == 0) {
    float v = redmax[0];
#pragma unroll
    for (int i = 1; i < 8; ++i) v = fmaxf(v, redmax[i]);
    redmax[0] = v;
  }
  __syncthreads();
  mx = redmax[0];

  float s = 0.f;
  for (int i = t; i < NPIX; i += 256) s += __expf(p[i] - mx);
#pragma unroll
  for (int o = 16; o > 0; o >>= 1) s += __shfl_xor(s, o, 32);
  if (lane == 0) redsum[w] = s;
  __syncthreads();
  if (t == 0) {
    float v = 0.f;
#pragma unroll
    for (int i = 0; i < 8; ++i) v += redsum[i];
    kstats[row * 2] = mx;
    kstats[row * 2 + 1] = 1.0f / v;
  }
}

// ---------------------------------------------------------------------------
// K3: ctx[b,h,d,e] = sum_n softmax(k)[d,n] * v_scaled[e,n]  (32x32, K=4096)
// softmax applied on the fly from kstats. 1 block (4 waves) per (b,h).
// ---------------------------------------------------------------------------
__global__ __launch_bounds__(128) void k_context(const float* __restrict__ qkv,
                                                 const float* __restrict__ kstats,
                                                 float* __restrict__ ctx) {
  const int bh = blockIdx.x;  // 0..63
  const int b = bh >> 2, h = bh & 3;
  const int t = threadIdx.x, wave = t >> 5, lane = t & 31;
  const int m = lane & 15, kh = lane >> 4;
  const int mi = wave >> 1, ni = wave & 1;
  const int drow = h * DHEAD + mi * 16 + m;  // k-row within batch (0..127)
  const float* arow = qkv + (size_t)b * NQKV * NPIX + (size_t)(HID + drow) * NPIX;
  const float* brow = qkv + (size_t)b * NQKV * NPIX +
                      (size_t)(2 * HID + h * DHEAD + ni * 16 + m) * NPIX;
  const float mx  = kstats[(b * HID + drow) * 2];
  const float rcp = kstats[(b * HID + drow) * 2 + 1];

  v8f acc = vzero8();
  for (int kk = 0; kk < NPIX; kk += 32) {
    __builtin_prefetch(arow + kk + 128, 0, 1);
    __builtin_prefetch(brow + kk + 128, 0, 1);
    v16h a = frag_g_row_sm(arow + kk, kh, mx, rcp);  // softmax(k) on the fly
    v16h bb = frag_g_row(brow + kk, kh);             // v row (pre-scaled by 1/n)
    acc = wmma_f16(a, bb, acc);
  }
  float* cb = ctx + (size_t)bh * DHEAD * DHEAD;
  const int mo = kh * 8;
#pragma unroll
  for (int g = 0; g < 8; ++g) {
    const int d = mi * 16 + g + mo;
    const int e = ni * 16 + m;
    cb[d * DHEAD + e] = acc[g];
  }
}

// ---------------------------------------------------------------------------
// K4 (fused): q softmax -> attn out = ctx^T @ q -> out = w_out @ attn + b_out
//             -> layernorm over channels * g.  1 block per (b, 64-pixel tile).
// LDS tiles stored fragment-major: [n][channel], K-contiguous per lane.
// ---------------------------------------------------------------------------
__global__ __launch_bounds__(256) void k_attn_out(const float* __restrict__ qkv,
                                                  const float* __restrict__ ctx,
                                                  const float* __restrict__ w_out,
                                                  const float* __restrict__ b_out,
                                                  const float* __restrict__ gln,
                                                  float* __restrict__ out) {
  __shared__ _Float16 qs[64][136];   // softmax(q)*scale, [n][d_global]
  __shared__ _Float16 as[64][136];   // attention output,  [n][e_global]
  __shared__ float colsum[64], colsq[64], colmean[64], colrstd[64];

  const int b = blockIdx.y;
  const int n0 = blockIdx.x * 64;
  const int t = threadIdx.x, wave = t >> 5, lane = t & 31;
  const int m = lane & 15, kh = lane >> 4;
  const int mo = kh * 8;
  const float* qb = qkv + (size_t)b * NQKV * NPIX;

  if (t < 64) { colsum[t] = 0.f; colsq[t] = 0.f; }

  // ---- phase 1: per-pixel softmax of q over d (32) per head ----
  {
    const int h = t >> 6, nl = t & 63;
    const float* qp = qb + (size_t)(h * DHEAD) * NPIX + n0 + nl;
    float qv[DHEAD];
    float mx = -3.4e38f;
#pragma unroll
    for (int d = 0; d < DHEAD; ++d) {
      qv[d] = qp[(size_t)d * NPIX];
      mx = fmaxf(mx, qv[d]);
    }
    float s = 0.f;
#pragma unroll
    for (int d = 0; d < DHEAD; ++d) { qv[d] = __expf(qv[d] - mx); s += qv[d]; }
    const float r = QSCALE / s;
#pragma unroll
    for (int d4 = 0; d4 < DHEAD; d4 += 4) {
      v4h hh;
#pragma unroll
      for (int j = 0; j < 4; ++j) hh[j] = (_Float16)(qv[d4 + j] * r);
      *(v4h*)&qs[nl][h * DHEAD + d4] = hh;  // packed b64 LDS store
    }
  }
  __syncthreads();

  // ---- phase 2: attn[e,n] = sum_d ctx[d,e] * q_soft[d,n], per head ----
  {
    const int h = wave >> 1, mt = wave & 1;
    const float* cb = ctx + (size_t)(b * NHEAD + h) * DHEAD * DHEAD;
    v16h a;  // A[m=e][k=d] = ctx[d][e]
    const int e = mt * 16 + m;
#pragma unroll
    for (int j = 0; j < 8; ++j) a[j] = (_Float16)cb[(kh * 8 + j) * DHEAD + e];
#pragma unroll
    for (int j = 0; j < 8; ++j) a[8 + j] = (_Float16)cb[(16 + kh * 8 + j) * DHEAD + e];
#pragma unroll
    for (int nt = 0; nt < 4; ++nt) {
      v16h bf = frag_lds_row(&qs[nt * 16 + m][h * DHEAD], kh);
      v8f acc = vzero8();
      acc = wmma_f16(a, bf, acc);
      // D rows e = mt*16 + kh*8 + (0..7): contiguous in as[n][.] -> packed stores
      v4h lo, hi;
#pragma unroll
      for (int j = 0; j < 4; ++j) { lo[j] = (_Float16)acc[j]; hi[j] = (_Float16)acc[4 + j]; }
      _Float16* dst = &as[nt * 16 + m][h * DHEAD + mt * 16 + mo];
      *(v4h*)dst = lo;
      *(v4h*)(dst + 4) = hi;
    }
  }
  __syncthreads();

  // ---- phase 3: out[o,n] = w_out[o,:] @ as[:,n]; wave owns 32 rows ----
  v8f cc[2][4];
#pragma unroll
  for (int i = 0; i < 2; ++i)
#pragma unroll
    for (int j = 0; j < 4; ++j) cc[i][j] = vzero8();

  for (int kc = 0; kc < HID; kc += 32) {
    v16h bf[4];
#pragma unroll
    for (int nt = 0; nt < 4; ++nt)
      bf[nt] = frag_lds_row(&as[nt * 16 + m][kc], kh);
#pragma unroll
    for (int mt2 = 0; mt2 < 2; ++mt2) {
      const float* wrow = w_out + (size_t)(wave * 32 + mt2 * 16 + m) * HID + kc;
      v16h a = frag_g_row(wrow, kh);
#pragma unroll
      for (int nt = 0; nt < 4; ++nt) cc[mt2][nt] = wmma_f16(a, bf[nt], cc[mt2][nt]);
    }
  }

  // ---- phase 4: add bias, accumulate per-column mean/var partials ----
#pragma unroll
  for (int mt2 = 0; mt2 < 2; ++mt2)
#pragma unroll
    for (int nt = 0; nt < 4; ++nt)
#pragma unroll
      for (int gi = 0; gi < 8; ++gi) {
        const int o = wave * 32 + mt2 * 16 + gi + mo;
        cc[mt2][nt][gi] += b_out[o];
      }

#pragma unroll
  for (int nt = 0; nt < 4; ++nt) {
    float s = 0.f, sq = 0.f;
#pragma unroll
    for (int mt2 = 0; mt2 < 2; ++mt2)
#pragma unroll
      for (int gi = 0; gi < 8; ++gi) {
        const float v = cc[mt2][nt][gi];
        s += v;
        sq += v * v;
      }
    // lane L and L+16 hold the same column n = nt*16 + (lane&15)
    s += __shfl_xor(s, 16, 32);
    sq += __shfl_xor(sq, 16, 32);
    if (lane < 16) {
      atomicAdd(&colsum[nt * 16 + lane], s);
      atomicAdd(&colsq[nt * 16 + lane], sq);
    }
  }
  __syncthreads();
  if (t < 64) {
    const float mean = colsum[t] * (1.0f / 256.0f);
    const float var = colsq[t] * (1.0f / 256.0f) - mean * mean;
    colmean[t] = mean;
    colrstd[t] = rsqrtf(var + LN_EPS);
  }
  __syncthreads();

  // ---- phase 5: normalize, scale by g, write ----
  float* ob = out + (size_t)b * CDIM * NPIX + n0;
#pragma unroll
  for (int mt2 = 0; mt2 < 2; ++mt2)
#pragma unroll
    for (int nt = 0; nt < 4; ++nt)
#pragma unroll
      for (int gi = 0; gi < 8; ++gi) {
        const int o = wave * 32 + mt2 * 16 + gi + mo;
        const int n = nt * 16 + m;
        ob[(size_t)o * NPIX + n] = (cc[mt2][nt][gi] - colmean[n]) * colrstd[n] * gln[o];
      }
}

// ---------------------------------------------------------------------------
extern "C" void kernel_launch(void* const* d_in, const int* in_sizes, int n_in,
                              void* d_out, int out_size, void* d_ws, size_t ws_size,
                              hipStream_t stream) {
  (void)in_sizes; (void)n_in; (void)out_size; (void)ws_size;
  const float* x     = (const float*)d_in[0];
  const float* w_qkv = (const float*)d_in[1];
  const float* w_out = (const float*)d_in[2];
  const float* b_out = (const float*)d_in[3];
  const float* g     = (const float*)d_in[4];
  float* out = (float*)d_out;

  float* qkv = (float*)d_ws;                               // 16*384*4096 f32
  float* ctx = qkv + (size_t)BATCH * NQKV * NPIX;          // 16*4*32*32 f32
  float* kst = ctx + (size_t)BATCH * NHEAD * DHEAD * DHEAD; // 2048*2 f32

  k_qkv_gemm<<<dim3(32, 3, 16), 256, 0, stream>>>(x, w_qkv, qkv);
  k_softmax_stats<<<dim3(2048), 256, 0, stream>>>(qkv, kst);
  k_context<<<dim3(64), 128, 0, stream>>>(qkv, kst, ctx);
  k_attn_out<<<dim3(64, 16), 256, 0, stream>>>(qkv, ctx, w_out, b_out, g, out);
}
